// SketchNet_46420006535342
// MI455X (gfx1250) — matrix operand order repaired and checked by reading, not compiled, not run on hardware
//
#include <hip/hip_runtime.h>
#include <hip/hip_bf16.h>
#include <math.h>

// ---------------------------------------------------------------------------
// Types for CDNA5 WMMA (gfx1250, wave32)
// ---------------------------------------------------------------------------
typedef __attribute__((ext_vector_type(16))) __bf16 v16bf;
typedef __attribute__((ext_vector_type(8)))  __bf16 v8bf;
typedef __attribute__((ext_vector_type(8)))  float  v8f;

#define DOUT 512
#define HDIM 1536
#define CLS  64

// ---------------------------------------------------------------------------
// fp32 -> bf16 conversion with row padding (rows >= n_rows zero-filled)
// ---------------------------------------------------------------------------
__global__ void f32_to_bf16_pad(const float* __restrict__ src, __bf16* __restrict__ dst,
                                int n_rows, int n_pad_rows, int K) {
  size_t i = (size_t)blockIdx.x * blockDim.x + threadIdx.x;
  size_t total = (size_t)n_pad_rows * K;
  if (i >= total) return;
  int row = (int)(i / (size_t)K);
  float v = (row < n_rows) ? src[i] : 0.0f;
  dst[i] = (__bf16)v;
}

// W [K, Dout] fp32  ->  Bt [Dout, K] bf16 (transposed for contiguous B-fragments)
__global__ void w_to_bf16_t(const float* __restrict__ W, __bf16* __restrict__ Bt,
                            int K, int Dout) {
  int i = blockIdx.x * blockDim.x + threadIdx.x;
  if (i >= K * Dout) return;
  int r = i / Dout, c = i % Dout;
  Bt[(size_t)c * K + r] = (__bf16)W[i];
}

// ---------------------------------------------------------------------------
// WMMA GEMM: C[Npad, Dout] = A[Npad, K](bf16) @ W  (W given as Bt[Dout,K] bf16)
// Block = 256 threads = 8 waves; wave computes 16 rows x 64 cols.
// ISA 7.12.2 layouts:
//   A 16x32 bf16: lane L -> row M=L&15, half=L>>4; elem i = K {half*16 + i} (contiguous, not ISA order)
//   B 32x16 bf16: lane L -> col N=L&15; elem i = K {half*16 + i} (same as A, not ISA order)
//   C 16x16 f32 : VGPR r -> row (half*8 + r), col (L&15)
// ---------------------------------------------------------------------------
__global__ __launch_bounds__(256)
void gemm_bf16_wmma(const __bf16* __restrict__ A,   // [Npad, K]
                    const __bf16* __restrict__ Bt,  // [Dout, K]
                    float* __restrict__ C,          // [Npad, Dout]
                    int K, int Dout) {
  const int wave = threadIdx.x >> 5;
  const int lane = threadIdx.x & 31;
  const int half = lane >> 4;
  const int l16  = lane & 15;
  const int Mbase = (blockIdx.y * 8 + wave) * 16;
  const int Nbase = blockIdx.x * 64;

  const __bf16* arow = A + (size_t)(Mbase + l16) * K;
  const __bf16* bcol = Bt + (size_t)(Nbase + l16) * K + half * 16;

  v8f c0 = {}, c1 = {}, c2 = {}, c3 = {};

  for (int kk = 0; kk < K; kk += 32) {
    union { v16bf v; v8bf h[2]; } a;
    a.h[0] = *(const v8bf*)(arow + kk + half * 16);       // K: kk + half*16 .. +7
    a.h[1] = *(const v8bf*)(arow + kk + half * 16 + 8);   // K: kk+half*16+8 .. +15

    v16bf b0 = *(const v16bf*)(bcol + kk);
    v16bf b1 = *(const v16bf*)(bcol + kk + (size_t)16 * K);
    v16bf b2 = *(const v16bf*)(bcol + kk + (size_t)32 * K);
    v16bf b3 = *(const v16bf*)(bcol + kk + (size_t)48 * K);

    c0 = __builtin_amdgcn_wmma_f32_16x16x32_bf16(false, a.v, false, b0, (short)0, c0, false, false);
    c1 = __builtin_amdgcn_wmma_f32_16x16x32_bf16(false, a.v, false, b1, (short)0, c1, false, false);
    c2 = __builtin_amdgcn_wmma_f32_16x16x32_bf16(false, a.v, false, b2, (short)0, c2, false, false);
    c3 = __builtin_amdgcn_wmma_f32_16x16x32_bf16(false, a.v, false, b3, (short)0, c3, false, false);
  }

  float* crow = C + (size_t)(Mbase + half * 8) * Dout + Nbase + l16;
#pragma unroll
  for (int r = 0; r < 8; ++r) {
    size_t o = (size_t)r * Dout;
    crow[o +  0] = c0[r];
    crow[o + 16] = c1[r];
    crow[o + 32] = c2[r];
    crow[o + 48] = c3[r];
  }
}

// ---------------------------------------------------------------------------
// Per-node init: segment-max sentinel (encoded 0 == "empty"), denom = 0
// ---------------------------------------------------------------------------
__global__ void init_node(unsigned* __restrict__ m_enc, float* __restrict__ denom, int N) {
  int i = blockIdx.x * blockDim.x + threadIdx.x;
  if (i >= N) return;
  m_enc[i] = 0u;
  denom[i] = 0.0f;
}

// out[n, d] = bias[d]   (scatter target pre-seeded with bias)
__global__ void init_out_bias(float* __restrict__ out, const float* __restrict__ bias,
                              size_t total, int D) {
  size_t i = (size_t)blockIdx.x * blockDim.x + threadIdx.x;
  if (i >= total) return;
  out[i] = bias[i % D];
}

// ---------------------------------------------------------------------------
// Edge phase 1: e = att . leaky_relu(xl[src] + xr[dst]); segment-max via
// order-preserving uint encoding + atomicMax. One wave per edge.
// ---------------------------------------------------------------------------
__global__ __launch_bounds__(256)
void edge_scores(const float* __restrict__ xl, const float* __restrict__ xr,
                 const float* __restrict__ att, const int* __restrict__ src,
                 const int* __restrict__ dst, float* __restrict__ e_val,
                 unsigned* __restrict__ m_enc, int E, int D) {
  int e = blockIdx.x * 8 + (threadIdx.x >> 5);
  int lane = threadIdx.x & 31;
  if (e >= E) return;
  int s = src[e], d = dst[e];
  const float* pl = xl + (size_t)s * D;
  const float* pr = xr + (size_t)d * D;
  float acc = 0.0f;
  for (int j = lane; j < D; j += 32) {
    float v = pl[j] + pr[j];
    v = (v > 0.0f) ? v : 0.2f * v;   // leaky_relu, NEG_SLOPE = 0.2
    acc += v * att[j];
  }
  for (int off = 16; off; off >>= 1) acc += __shfl_down(acc, off, 32);
  if (lane == 0) {
    e_val[e] = acc;
    unsigned k = __float_as_uint(acc);
    k = (k & 0x80000000u) ? ~k : (k | 0x80000000u);
    atomicMax(m_enc + d, k);
  }
}

// decode segment-max; empty / non-finite -> 0 (matches reference isfinite guard)
__global__ void decode_m(const unsigned* __restrict__ m_enc, float* __restrict__ m, int N) {
  int i = blockIdx.x * blockDim.x + threadIdx.x;
  if (i >= N) return;
  unsigned k = m_enc[i];
  float v = 0.0f;
  if (k != 0u) {
    unsigned b = (k & 0x80000000u) ? (k ^ 0x80000000u) : ~k;
    v = __uint_as_float(b);
    if (!isfinite(v)) v = 0.0f;
  }
  m[i] = v;
}

// Edge phase 2: ex = exp(e - m[dst]); denom[dst] += ex
__global__ void edge_exp(const float* __restrict__ e_val, const float* __restrict__ m,
                         const int* __restrict__ dst, float* __restrict__ ex,
                         float* __restrict__ denom, int E) {
  int e = blockIdx.x * blockDim.x + threadIdx.x;
  if (e >= E) return;
  int d = dst[e];
  float v = __expf(e_val[e] - m[d]);
  ex[e] = v;
  atomicAdd(denom + d, v);
}

// Edge phase 3: out[dst] += (ex/denom[dst]) * xl[src]. One wave per edge.
__global__ __launch_bounds__(256)
void edge_scatter(const float* __restrict__ xl, const float* __restrict__ ex,
                  const float* __restrict__ denom, const int* __restrict__ src,
                  const int* __restrict__ dst, float* __restrict__ out, int E, int D) {
  int e = blockIdx.x * 8 + (threadIdx.x >> 5);
  int lane = threadIdx.x & 31;
  if (e >= E) return;
  int s = src[e], d = dst[e];
  float alpha = ex[e] / (denom[d] + 1e-16f);
  const float* pl = xl + (size_t)s * D;
  float* po = out + (size_t)d * D;
  for (int j = lane; j < D; j += 32) atomicAdd(po + j, alpha * pl[j]);
}

__global__ void relu_k(float* __restrict__ h, size_t n) {
  size_t i = (size_t)blockIdx.x * blockDim.x + threadIdx.x;
  if (i >= n) return;
  h[i] = fmaxf(h[i], 0.0f);
}

__global__ void zero_f(float* __restrict__ p, int n) {
  int i = blockIdx.x * blockDim.x + threadIdx.x;
  if (i < n) p[i] = 0.0f;
}

__global__ void scale_f(float* __restrict__ p, int n, float s) {
  int i = blockIdx.x * blockDim.x + threadIdx.x;
  if (i < n) p[i] *= s;
}

// column sums (mean pool) with row strips + single atomic per (thread,strip)
__global__ __launch_bounds__(256)
void col_sum(const float* __restrict__ h, float* __restrict__ emb, int N, int D, int strips) {
  int d = blockIdx.x * 256 + threadIdx.x;
  if (d >= D) return;
  int s = blockIdx.y;
  int r0 = (int)((long long)N * s / strips);
  int r1 = (int)((long long)N * (s + 1) / strips);
  float acc = 0.0f;
  for (int n = r0; n < r1; ++n) acc += h[(size_t)n * D + d];
  atomicAdd(&emb[d], acc);
}

// dense fc: out[j] = (relu?) b[j] + sum_i in[i]*W[i*Out + j]
__global__ void fc(const float* __restrict__ in, const float* __restrict__ W,
                   const float* __restrict__ b, float* __restrict__ out,
                   int In, int Out, int do_relu) {
  int j = blockIdx.x * blockDim.x + threadIdx.x;
  if (j >= Out) return;
  float acc = b[j];
  for (int i = 0; i < In; ++i) acc += in[i] * W[(size_t)i * Out + j];
  out[j] = do_relu ? fmaxf(acc, 0.0f) : acc;
}

__global__ void log_softmax64(const float* __restrict__ in, float* __restrict__ out, int C) {
  __shared__ float sh[64];
  int t = threadIdx.x;
  float v = (t < C) ? in[t] : -INFINITY;
  sh[t] = v; __syncthreads();
  for (int o = 32; o; o >>= 1) { if (t < o) sh[t] = fmaxf(sh[t], sh[t + o]); __syncthreads(); }
  float mx = sh[0]; __syncthreads();
  float ev = (t < C) ? __expf(v - mx) : 0.0f;
  sh[t] = ev; __syncthreads();
  for (int o = 32; o; o >>= 1) { if (t < o) sh[t] += sh[t + o]; __syncthreads(); }
  float lse = __logf(sh[0]);
  if (t < C) out[t] = v - mx - lse;
}

// ---------------------------------------------------------------------------
// Host-side orchestration
// ---------------------------------------------------------------------------
static inline int cdiv(long long a, long long b) { return (int)((a + b - 1) / b); }

struct Scratch {
  float*   hA;     // [Npad, 512] feature ping buffer (also scatter output)
  float*   xl;     // [Npad, 512]
  float*   xr;     // [Npad, 512]
  __bf16*  Xbf;    // [Npad, 512] bf16 input staging
  __bf16*  WlT;    // [512, 512] bf16 (transposed)
  __bf16*  WrT;
  float*   e_val;  // [E]
  float*   exv;    // [E]
  unsigned* m_enc; // [N]
  float*   m_dec;  // [N]
  float*   denom;  // [N]
  float*   emb;    // [1536]
  float*   t1; float* t2; float* t3;
};

static void run_gat_layer(hipStream_t st, const float* x_in, int Din, int enc,
                          const float* Wl_all, const float* Wr_all,
                          const float* att_all, const float* b_all,
                          const int* src, const int* dst,
                          int N, int Npad, int E, bool relu, const Scratch& S) {
  const int D = DOUT;
  const float* Wl  = Wl_all  + (size_t)enc * Din * D;
  const float* Wr  = Wr_all  + (size_t)enc * Din * D;
  const float* att = att_all + (size_t)enc * D;
  const float* bias = b_all  + (size_t)enc * D;

  // 1) convert input features to padded bf16
  f32_to_bf16_pad<<<cdiv((long long)Npad * Din, 256), 256, 0, st>>>(x_in, S.Xbf, N, Npad, Din);
  // 2) convert + transpose weights to bf16
  w_to_bf16_t<<<cdiv((long long)Din * D, 256), 256, 0, st>>>(Wl, S.WlT, Din, D);
  w_to_bf16_t<<<cdiv((long long)Din * D, 256), 256, 0, st>>>(Wr, S.WrT, Din, D);
  // 3) WMMA GEMMs: xl = X @ Wl ; xr = X @ Wr
  dim3 gg(D / 64, Npad / 128);
  gemm_bf16_wmma<<<gg, 256, 0, st>>>(S.Xbf, S.WlT, S.xl, Din, D);
  gemm_bf16_wmma<<<gg, 256, 0, st>>>(S.Xbf, S.WrT, S.xr, Din, D);
  // 4) per-node init
  init_node<<<cdiv(N, 256), 256, 0, st>>>(S.m_enc, S.denom, N);
  init_out_bias<<<cdiv((long long)Npad * D, 256), 256, 0, st>>>(S.hA, bias, (size_t)Npad * D, D);
  // 5) attention logits + segment max
  edge_scores<<<cdiv(E, 8), 256, 0, st>>>(S.xl, S.xr, att, src, dst, S.e_val, S.m_enc, E, D);
  decode_m<<<cdiv(N, 256), 256, 0, st>>>(S.m_enc, S.m_dec, N);
  // 6) softmax numerator + denominator
  edge_exp<<<cdiv(E, 256), 256, 0, st>>>(S.e_val, S.m_dec, dst, S.exv, S.denom, E);
  // 7) weighted message scatter
  edge_scatter<<<cdiv(E, 8), 256, 0, st>>>(S.xl, S.exv, S.denom, src, dst, S.hA, E, D);
  // 8) activation
  if (relu) relu_k<<<cdiv((long long)Npad * D, 256), 256, 0, st>>>(S.hA, (size_t)Npad * D);
}

extern "C" void kernel_launch(void* const* d_in, const int* in_sizes, int n_in,
                              void* d_out, int out_size, void* d_ws, size_t ws_size,
                              hipStream_t stream) {
  (void)n_in; (void)out_size; (void)ws_size;
  const int D_IN = 256, D = DOUT;
  const int N = in_sizes[0] / D_IN;
  const int E = in_sizes[1] / 2;
  const int Npad = ((N + 127) / 128) * 128;

  // --- inputs (setup_inputs dict order, lists flattened) ---
  const float* xs[3]  = { (const float*)d_in[0], (const float*)d_in[2], (const float*)d_in[4] };
  const int*   eis[3] = { (const int*)d_in[1],   (const int*)d_in[3],   (const int*)d_in[5] };
  const float* WlL[4]  = { (const float*)d_in[6],  (const float*)d_in[7],  (const float*)d_in[8],  (const float*)d_in[9]  };
  const float* WrL[4]  = { (const float*)d_in[10], (const float*)d_in[11], (const float*)d_in[12], (const float*)d_in[13] };
  const float* attL[4] = { (const float*)d_in[14], (const float*)d_in[15], (const float*)d_in[16], (const float*)d_in[17] };
  const float* bL[4]   = { (const float*)d_in[18], (const float*)d_in[19], (const float*)d_in[20], (const float*)d_in[21] };
  const float* lin1_w = (const float*)d_in[22]; const float* lin1_b = (const float*)d_in[23];
  const float* lin2_w = (const float*)d_in[24]; const float* lin2_b = (const float*)d_in[25];
  const float* lin3_w = (const float*)d_in[26]; const float* lin3_b = (const float*)d_in[27];

  // --- carve scratch (256B-aligned slabs) ---
  char* p = (char*)d_ws;
  auto alloc = [&](size_t bytes) -> void* {
    void* r = (void*)p;
    p += (bytes + 255) & ~(size_t)255;
    return r;
  };
  Scratch S;
  S.hA    = (float*)alloc((size_t)Npad * D * 4);
  S.xl    = (float*)alloc((size_t)Npad * D * 4);
  S.xr    = (float*)alloc((size_t)Npad * D * 4);
  S.Xbf   = (__bf16*)alloc((size_t)Npad * D * 2);
  S.WlT   = (__bf16*)alloc((size_t)D * D * 2);
  S.WrT   = (__bf16*)alloc((size_t)D * D * 2);
  S.e_val = (float*)alloc((size_t)E * 4);
  S.exv   = (float*)alloc((size_t)E * 4);
  S.m_enc = (unsigned*)alloc((size_t)N * 4);
  S.m_dec = (float*)alloc((size_t)N * 4);
  S.denom = (float*)alloc((size_t)N * 4);
  S.emb   = (float*)alloc(HDIM * 4);
  S.t1    = (float*)alloc(HDIM * 4);
  S.t2    = (float*)alloc(HDIM * 4);
  S.t3    = (float*)alloc(CLS * 4);

  zero_f<<<cdiv(HDIM, 256), 256, 0, stream>>>(S.emb, HDIM);

  // concat order: [sk (enc 2), lhs (enc 0), rhs (enc 1)]
  const int enc_order[3] = { 2, 0, 1 };
  const int emb_off[3]   = { 0, D, 2 * D };
  const int layers[5]    = { 0, 1, 1, 2, 3 };  // conv index 1 applied twice

  for (int k = 0; k < 3; ++k) {
    const int enc = enc_order[k];
    const int* src = eis[enc];
    const int* dst = eis[enc] + E;
    const float* cur = xs[enc];
    int Din = D_IN;
    for (int li = 0; li < 5; ++li) {
      const int L = layers[li];
      run_gat_layer(stream, cur, Din, enc, WlL[L], WrL[L], attL[L], bL[L],
                    src, dst, N, Npad, E, /*relu=*/li < 4, S);
      cur = S.hA;
      Din = D;
    }
    // global mean pool (accumulate sums; scaled below)
    dim3 cg(D / 256, 64);
    col_sum<<<cg, 256, 0, stream>>>(S.hA, S.emb + emb_off[k], N, D, 64);
  }

  scale_f<<<cdiv(HDIM, 256), 256, 0, stream>>>(S.emb, HDIM, 1.0f / (float)N);

  // MLP head + log_softmax
  fc<<<cdiv(HDIM, 256), 256, 0, stream>>>(S.emb, lin1_w, lin1_b, S.t1, HDIM, HDIM, 1);
  fc<<<cdiv(HDIM, 256), 256, 0, stream>>>(S.t1, lin2_w, lin2_b, S.t2, HDIM, HDIM, 1);
  fc<<<cdiv(CLS, 256), 256, 0, stream>>>(S.t2, lin3_w, lin3_b, S.t3, HDIM, CLS, 0);
  log_softmax64<<<1, 64, 0, stream>>>(S.t3, (float*)d_out, CLS);
}
